// MoE_Adapter_60421599920597
// MI455X (gfx1250) — matrix operand order repaired and checked by reading, not compiled, not run on hardware
//
#include <hip/hip_runtime.h>
#include <stdint.h>

#define BB 4
#define SS 4096
#define DD 2048
#define EE 6
#define RR 16
#define BS (BB*SS)      /* 16384 tokens */
#define NCAT 128        /* 112 low-rank cols padded to 128 */

typedef __attribute__((ext_vector_type(4)))  float          f32x4;
typedef __attribute__((ext_vector_type(4)))  int            v4i;
typedef __attribute__((ext_vector_type(8)))  float          v8f;
typedef __attribute__((ext_vector_type(16))) __bf16         v16bf;
typedef __attribute__((ext_vector_type(16))) unsigned short v16u;
typedef __attribute__((ext_vector_type(8)))  unsigned short v8u;

__device__ __forceinline__ unsigned short bf16_rne(float f) {
  unsigned u = __float_as_uint(f);
  u += 0x7FFFu + ((u >> 16) & 1u);           // round-to-nearest-even
  return (unsigned short)(u >> 16);
}
__device__ __forceinline__ float bf16_f32(unsigned short h) {
  return __uint_as_float(((unsigned)h) << 16);
}

// -------- gfx1250 async global->LDS copy (guarded; fallback = sync copy) ----
#if defined(__has_builtin)
#if __has_builtin(__builtin_amdgcn_global_load_async_to_lds_b128)
#define HAVE_ASYNC_LDS 1
#endif
#endif

__device__ __forceinline__ void cp_b128(const unsigned short* g, unsigned short* l) {
#ifdef HAVE_ASYNC_LDS
  void* gv = (void*)g;
  void* lv = (void*)l;
  __builtin_amdgcn_global_load_async_to_lds_b128(
      (__attribute__((address_space(1))) v4i*)gv,
      (__attribute__((address_space(3))) v4i*)lv, 0, 0);
#else
  *(v8u*)l = *(const v8u*)g;
#endif
}
__device__ __forceinline__ void wait_async_copies() {
#ifdef HAVE_ASYNC_LDS
#if __has_builtin(__builtin_amdgcn_s_wait_asynccnt)
  __builtin_amdgcn_s_wait_asynccnt(0);
#else
  asm volatile("s_wait_asynccnt 0x0" ::: "memory");
#endif
#endif
}

// ---------------------------------------------------------------------------
// Router: one wave32 per token. logits = x . W_router + b; softmax; top-2
// scatter into w_per_e[m][0..7] (E=6 used, padded to 8).
// ---------------------------------------------------------------------------
__global__ void router_kernel(const float* __restrict__ x,
                              const float* __restrict__ Wr,
                              const float* __restrict__ br,
                              float* __restrict__ w_per_e) {
  const int wave = threadIdx.x >> 5;
  const int lane = threadIdx.x & 31;
  const int m = blockIdx.x * 8 + wave;
  const float* xr = x + (size_t)m * DD;
  float acc[EE] = {0.f, 0.f, 0.f, 0.f, 0.f, 0.f};
  for (int k = lane; k < DD; k += 32) {
    const float xv = xr[k];
    const float* wr = Wr + (size_t)k * EE;
#pragma unroll
    for (int e = 0; e < EE; ++e) acc[e] += xv * wr[e];
  }
#pragma unroll
  for (int off = 16; off > 0; off >>= 1) {
#pragma unroll
    for (int e = 0; e < EE; ++e) acc[e] += __shfl_down(acc[e], off, 32);
  }
  if (lane == 0) {
    float g[EE];
    float mx = -1e30f;
#pragma unroll
    for (int e = 0; e < EE; ++e) { g[e] = acc[e] + br[e]; mx = fmaxf(mx, g[e]); }
    float s = 0.f;
#pragma unroll
    for (int e = 0; e < EE; ++e) { g[e] = __expf(g[e] - mx); s += g[e]; }
    const float inv = 1.0f / s;
#pragma unroll
    for (int e = 0; e < EE; ++e) g[e] *= inv;
    int i1 = 0; float g1 = g[0];
#pragma unroll
    for (int e = 1; e < EE; ++e) if (g[e] > g1) { g1 = g[e]; i1 = e; }
    int i2 = -1; float g2 = -1e30f;
#pragma unroll
    for (int e = 0; e < EE; ++e) if (e != i1 && g[e] > g2) { g2 = g[e]; i2 = e; }
    float* wo = w_per_e + (size_t)m * 8;
#pragma unroll
    for (int e = 0; e < 8; ++e) wo[e] = (e == i1) ? g1 : ((e == i2) ? g2 : 0.f);
  }
}

// ---------------------------------------------------------------------------
// Small packing kernels (fallback path, f32 concatenated low-rank weights)
// ---------------------------------------------------------------------------
__global__ void build_acat(const float* __restrict__ A_s,
                           const float* __restrict__ A_r,
                           float* __restrict__ Acat) {
  const int idx = blockIdx.x * blockDim.x + threadIdx.x; // 2048*128
  const int k = idx >> 7, c = idx & 127;
  float v = 0.f;
  if (c < RR) {
    v = A_s[(size_t)k * RR + c];
  } else if (c < 16 + EE * RR) {
    const int e = (c - RR) >> 4, r = (c - RR) & 15;
    v = A_r[((size_t)e * DD + k) * RR + r];
  }
  Acat[idx] = v;
}

__global__ void build_bcat(const float* __restrict__ B_s,
                           const float* __restrict__ B_r,
                           float* __restrict__ Bcat) {
  const int idx = blockIdx.x * blockDim.x + threadIdx.x; // 128*2048
  const int j = idx >> 11, n = idx & 2047;
  float v = 0.f;
  if (j < RR) v = B_s[(size_t)j * DD + n];
  else if (j < 16 + EE * RR) v = B_r[(size_t)(j - RR) * DD + n];
  Bcat[idx] = v;
}

// Fold routed gate weights into hcat columns 16..111 (shared cols 0..15 untouched).
__global__ void fold_gates(float* __restrict__ hcat,
                           const float* __restrict__ w_per_e) {
  const int idx = blockIdx.x * blockDim.x + threadIdx.x; // BS*128
  const int m = idx >> 7, c = idx & 127;
  if (c >= RR && c < 16 + EE * RR) {
    hcat[idx] *= w_per_e[(size_t)m * 8 + ((c - RR) >> 4)];
  }
}

// ---------------------------------------------------------------------------
// Pre-split kernels (fast path): f32 -> {bf16 hi, bf16 lo}, done ONCE.
// ---------------------------------------------------------------------------
__global__ void split_f32_bf16(const float* __restrict__ in,
                               unsigned short* __restrict__ hi,
                               unsigned short* __restrict__ lo) {
  const int i = blockIdx.x * blockDim.x + threadIdx.x;
  const float f = in[i];
  const unsigned short h = bf16_rne(f);
  hi[i] = h;
  lo[i] = bf16_rne(f - bf16_f32(h));
}

// Transpose+split W_base [K=2048][N=2048] -> Wt_{hi,lo} [N][K] bf16.
__global__ void transpose_split_2048(const float* __restrict__ W,
                                     unsigned short* __restrict__ Th,
                                     unsigned short* __restrict__ Tl) {
  __shared__ unsigned short sh[64][72], sl[64][72];
  const int tid = threadIdx.x;
  const int kb = blockIdx.x * 64;
  const int nb = blockIdx.y * 64;
#pragma unroll
  for (int i = 0; i < 4; ++i) {
    const int lin = tid + i * 256;     // 1024 float4 = 64 rows * 16
    const int r = lin >> 4;            // k local
    const int c4 = (lin & 15) << 2;    // n local
    f32x4 v = *(const f32x4*)(W + (size_t)(kb + r) * DD + nb + c4);
#pragma unroll
    for (int j = 0; j < 4; ++j) {
      const unsigned short h = bf16_rne(v[j]);
      sh[r][c4 + j] = h;
      sl[r][c4 + j] = bf16_rne(v[j] - bf16_f32(h));
    }
  }
  __syncthreads();
#pragma unroll
  for (int i = 0; i < 2; ++i) {
    const int lin = tid + i * 256;     // 512 chunks = 64 n-rows * 8
    const int nl = lin >> 3;
    const int q = (lin & 7) << 3;      // k start
    v8u ph, pl;
#pragma unroll
    for (int j = 0; j < 8; ++j) { ph[j] = sh[q + j][nl]; pl[j] = sl[q + j][nl]; }
    *(v8u*)(Th + (size_t)(nb + nl) * DD + kb + q) = ph;
    *(v8u*)(Tl + (size_t)(nb + nl) * DD + kb + q) = pl;
  }
}

// Acat^T split: out [c=128][k=2048] bf16 hi/lo built straight from A_s/A_r.
__global__ void build_acatt_split(const float* __restrict__ A_s,
                                  const float* __restrict__ A_r,
                                  unsigned short* __restrict__ hi,
                                  unsigned short* __restrict__ lo) {
  const int idx = blockIdx.x * blockDim.x + threadIdx.x; // 128*2048
  const int c = idx >> 11, k = idx & 2047;
  float v = 0.f;
  if (c < RR) {
    v = A_s[(size_t)k * RR + c];
  } else if (c < 16 + EE * RR) {
    const int e = (c - RR) >> 4, r = (c - RR) & 15;
    v = A_r[((size_t)e * DD + k) * RR + r];
  }
  const unsigned short h = bf16_rne(v);
  hi[idx] = h;
  lo[idx] = bf16_rne(v - bf16_f32(h));
}

// Bcat^T split: out [n=2048][j=128] bf16 hi/lo built straight from B_s/B_r.
__global__ void build_bct_split(const float* __restrict__ B_s,
                                const float* __restrict__ B_r,
                                unsigned short* __restrict__ hi,
                                unsigned short* __restrict__ lo) {
  const int idx = blockIdx.x * blockDim.x + threadIdx.x; // 2048*128
  const int n = idx >> 7, j = idx & 127;
  float v = 0.f;
  if (j < RR) v = B_s[(size_t)j * DD + n];
  else if (j < 16 + EE * RR) v = B_r[(size_t)(j - RR) * DD + n];
  const unsigned short h = bf16_rne(v);
  hi[idx] = h;
  lo[idx] = bf16_rne(v - bf16_f32(h));
}

// ---------------------------------------------------------------------------
// Common GEMM geometry
// ---------------------------------------------------------------------------
#define BM 128
#define BN 128
#define BKK 32
#define LDSP 40   /* ushort row pitch: 80B rows, 16B-aligned, conflict-spread */

// ---------------------------------------------------------------------------
// FAST PATH: bf16x3 GEMM with pre-split bf16 inputs; A row-major [M][K],
// B pre-transposed [N][K]; DOUBLE-BUFFERED async global->LDS staging so the
// next chunk's DMA flies under the current chunk's 24 WMMAs.
// ---------------------------------------------------------------------------
__global__ __launch_bounds__(256)
void gemm_bf16_direct(const unsigned short* __restrict__ Ahi,
                      const unsigned short* __restrict__ Alo,
                      const unsigned short* __restrict__ Bhi,
                      const unsigned short* __restrict__ Blo, int K,
                      const unsigned short* __restrict__ A2hi,
                      const unsigned short* __restrict__ A2lo,
                      const unsigned short* __restrict__ B2hi,
                      const unsigned short* __restrict__ B2lo, int K2,
                      const float* __restrict__ bias,
                      float* __restrict__ C, int ldc) {
  // SH[buf][0]=A_hi [1]=A_lo [2]=B_hi [3]=B_lo  (2 x 40KB = 80KB of 320KB WGP LDS)
  __shared__ unsigned short SH[2][4][BM][LDSP];

  const int tid   = threadIdx.x;
  const int lane  = tid & 31;
  const int wid   = tid >> 5;
  const int waveM = wid & 1;
  const int waveN = wid >> 1;
  const int half  = lane >> 4;
  const int lm    = lane & 15;

  v8f acc[8] = {};

  const int m0 = blockIdx.x * BM;
  const int n0 = blockIdx.y * BN;

  const int nc1 = K / BKK;
  const int nc2 = K2 / BKK;
  const int nct = nc1 + nc2;

  // stage chunk c into LDS buffer buf (uniform pointer math; 32-bit voffset
  // so the backend can use the scalar-base SADDR async encoding)
  auto stage_chunk = [&](int c, int buf) {
    const unsigned short *ah, *al, *bh, *bl;
    int pK, kb;
    if (c < nc1) { ah = Ahi;  al = Alo;  bh = Bhi;  bl = Blo;  pK = K;  kb = c * BKK; }
    else         { ah = A2hi; al = A2lo; bh = B2hi; bl = B2lo; pK = K2; kb = (c - nc1) * BKK; }
    const unsigned short* gb[4] = {
      ah + (size_t)m0 * pK + kb,
      al + (size_t)m0 * pK + kb,
      bh + (size_t)n0 * pK + kb,
      bl + (size_t)n0 * pK + kb };
#pragma unroll
    for (int t4 = 0; t4 < 4; ++t4) {
#pragma unroll
      for (int i = 0; i < 2; ++i) {
        const int ll = tid + i * 256;                 // 512 transfers per tile
        const int r = ll >> 2;                        // 0..127 row
        const int q = (ll & 3) << 3;                  // 0,8,16,24 elements
        const unsigned off32 = (unsigned)r * (unsigned)pK + (unsigned)q;
        cp_b128(gb[t4] + off32, &SH[buf][t4][r][q]);
      }
    }
  };

  if (nct > 0) stage_chunk(0, 0);

  for (int c = 0; c < nct; ++c) {
    const int buf = c & 1;
    wait_async_copies();       // chunk c resident in SH[buf]
    __syncthreads();           // all waves done reading SH[buf^1] (iter c-1)
    if (c + 1 < nct) stage_chunk(c + 1, buf ^ 1);   // DMA under compute

    // ---- B fragments: elements 0..15 = K 16*half .. 16*half+15 ----
    v16bf bhi[2], blo[2];
#pragma unroll
    for (int nt = 0; nt < 2; ++nt) {
      const int n = waveN * 32 + nt * 16 + lm;
      v8u h0 = *(const v8u*)&SH[buf][2][n][half * 16];
      v8u h1 = *(const v8u*)&SH[buf][2][n][half * 16 + 8];
      v8u l0 = *(const v8u*)&SH[buf][3][n][half * 16];
      v8u l1 = *(const v8u*)&SH[buf][3][n][half * 16 + 8];
      v16u th, tl;
#pragma unroll
      for (int j = 0; j < 8; ++j) {
        th[j] = h0[j]; th[8 + j] = h1[j];
        tl[j] = l0[j]; tl[8 + j] = l1[j];
      }
      bhi[nt] = __builtin_bit_cast(v16bf, th);
      blo[nt] = __builtin_bit_cast(v16bf, tl);
    }
    // ---- A fragments: elements 0..15 = K 16*half .. 16*half+15 ----
#pragma unroll
    for (int mt = 0; mt < 4; ++mt) {
      const int m = waveM * 64 + mt * 16 + lm;
      v8u h0 = *(const v8u*)&SH[buf][0][m][half * 16];
      v8u h1 = *(const v8u*)&SH[buf][0][m][half * 16 + 8];
      v8u l0 = *(const v8u*)&SH[buf][1][m][half * 16];
      v8u l1 = *(const v8u*)&SH[buf][1][m][half * 16 + 8];
      v16u th, tl;
#pragma unroll
      for (int j = 0; j < 8; ++j) {
        th[j] = h0[j]; th[8 + j] = h1[j];
        tl[j] = l0[j]; tl[8 + j] = l1[j];
      }
      v16bf ahi = __builtin_bit_cast(v16bf, th);
      v16bf alo = __builtin_bit_cast(v16bf, tl);
#pragma unroll
      for (int nt = 0; nt < 2; ++nt) {
        v8f cc = acc[mt * 2 + nt];
        cc = __builtin_amdgcn_wmma_f32_16x16x32_bf16(false, ahi, false, blo[nt], (short)0, cc, false, false);
        cc = __builtin_amdgcn_wmma_f32_16x16x32_bf16(false, alo, false, bhi[nt], (short)0, cc, false, false);
        cc = __builtin_amdgcn_wmma_f32_16x16x32_bf16(false, ahi, false, bhi[nt], (short)0, cc, false, false);
        acc[mt * 2 + nt] = cc;
      }
    }
    // no trailing barrier: next iteration's wait+barrier covers the hazard
  }

#pragma unroll
  for (int mt = 0; mt < 4; ++mt) {
#pragma unroll
    for (int nt = 0; nt < 2; ++nt) {
      const int col = n0 + waveN * 32 + nt * 16 + lm;
      const float bsv = bias ? bias[col] : 0.0f;
      const int rowbase = m0 + waveM * 64 + mt * 16 + half * 8;
#pragma unroll
      for (int r = 0; r < 8; ++r) {
        C[(size_t)(rowbase + r) * ldc + col] = acc[mt * 2 + nt][r] + bsv;
      }
    }
  }
}

// ---------------------------------------------------------------------------
// FALLBACK PATH: bf16x3 GEMM, f32 inputs, split done at LDS staging
// (used when ws_size can't hold the pre-split arrays).
// ---------------------------------------------------------------------------
__global__ __launch_bounds__(256)
void gemm_bf16x3(const float* __restrict__ A, int lda,
                 const float* __restrict__ B, int ldb, int K,
                 const float* __restrict__ A2, int lda2,
                 const float* __restrict__ B2, int ldb2, int K2,
                 const float* __restrict__ bias,
                 float* __restrict__ C, int ldc) {
  __shared__ unsigned short Ah[BM][LDSP], Al[BM][LDSP];
  __shared__ unsigned short Bh[BN][LDSP], Bl[BN][LDSP];

  const int tid   = threadIdx.x;
  const int lane  = tid & 31;
  const int wid   = tid >> 5;
  const int waveM = wid & 1;
  const int waveN = wid >> 1;
  const int half  = lane >> 4;
  const int lm    = lane & 15;

  v8f acc[8] = {};

  const int m0 = blockIdx.x * BM;
  const int n0 = blockIdx.y * BN;

  for (int phase = 0; phase < 2; ++phase) {
    const float* pA  = phase ? A2   : A;
    const float* pB  = phase ? B2   : B;
    const int   plda = phase ? lda2 : lda;
    const int   pldb = phase ? ldb2 : ldb;
    const int   pK   = phase ? K2   : K;

    for (int kb = 0; kb < pK; kb += BKK) {
      {
        const float* Ab = pA + (size_t)m0 * plda + kb;
#pragma unroll
        for (int i = 0; i < 4; ++i) {
          const int lin = tid + i * 256;
          const int row = lin >> 3;
          const int c4  = (lin & 7) << 2;
          f32x4 v = *(const f32x4*)(Ab + (size_t)row * plda + c4);
#pragma unroll
          for (int j = 0; j < 4; ++j) {
            const unsigned short h = bf16_rne(v[j]);
            Ah[row][c4 + j] = h;
            Al[row][c4 + j] = bf16_rne(v[j] - bf16_f32(h));
          }
        }
        const float* Bb = pB + (size_t)kb * pldb + n0;
#pragma unroll
        for (int i = 0; i < 4; ++i) {
          const int lin = tid + i * 256;
          const int kr  = lin >> 5;
          const int c4  = (lin & 31) << 2;
          f32x4 v = *(const f32x4*)(Bb + (size_t)kr * pldb + c4);
#pragma unroll
          for (int j = 0; j < 4; ++j) {
            const unsigned short h = bf16_rne(v[j]);
            Bh[c4 + j][kr] = h;
            Bl[c4 + j][kr] = bf16_rne(v[j] - bf16_f32(h));
          }
        }
      }
      __syncthreads();

      v16bf bhi[2], blo[2];
#pragma unroll
      for (int nt = 0; nt < 2; ++nt) {
        const int n = waveN * 32 + nt * 16 + lm;
        v8u h0 = *(const v8u*)&Bh[n][half * 16];
        v8u h1 = *(const v8u*)&Bh[n][half * 16 + 8];
        v8u l0 = *(const v8u*)&Bl[n][half * 16];
        v8u l1 = *(const v8u*)&Bl[n][half * 16 + 8];
        v16u th, tl;
#pragma unroll
        for (int j = 0; j < 8; ++j) {
          th[j] = h0[j]; th[8 + j] = h1[j];
          tl[j] = l0[j]; tl[8 + j] = l1[j];
        }
        bhi[nt] = __builtin_bit_cast(v16bf, th);
        blo[nt] = __builtin_bit_cast(v16bf, tl);
      }
#pragma unroll
      for (int mt = 0; mt < 4; ++mt) {
        const int m = waveM * 64 + mt * 16 + lm;
        v8u h0 = *(const v8u*)&Ah[m][half * 16];
        v8u h1 = *(const v8u*)&Ah[m][half * 16 + 8];
        v8u l0 = *(const v8u*)&Al[m][half * 16];
        v8u l1 = *(const v8u*)&Al[m][half * 16 + 8];
        v16u th, tl;
#pragma unroll
        for (int j = 0; j < 8; ++j) {
          th[j] = h0[j]; th[8 + j] = h1[j];
          tl[j] = l0[j]; tl[8 + j] = l1[j];
        }
        v16bf ahi = __builtin_bit_cast(v16bf, th);
        v16bf alo = __builtin_bit_cast(v16bf, tl);
#pragma unroll
        for (int nt = 0; nt < 2; ++nt) {
          v8f c = acc[mt * 2 + nt];
          c = __builtin_amdgcn_wmma_f32_16x16x32_bf16(false, ahi, false, blo[nt], (short)0, c, false, false);
          c = __builtin_amdgcn_wmma_f32_16x16x32_bf16(false, alo, false, bhi[nt], (short)0, c, false, false);
          c = __builtin_amdgcn_wmma_f32_16x16x32_bf16(false, ahi, false, bhi[nt], (short)0, c, false, false);
          acc[mt * 2 + nt] = c;
        }
      }
      __syncthreads();
    }
  }

#pragma unroll
  for (int mt = 0; mt < 4; ++mt) {
#pragma unroll
    for (int nt = 0; nt < 2; ++nt) {
      const int col = n0 + waveN * 32 + nt * 16 + lm;
      const float bsv = bias ? bias[col] : 0.0f;
      const int rowbase = m0 + waveM * 64 + mt * 16 + half * 8;
#pragma unroll
      for (int r = 0; r < 8; ++r) {
        C[(size_t)(rowbase + r) * ldc + col] = acc[mt * 2 + nt][r] + bsv;
      }
    }
  }
}

// ---------------------------------------------------------------------------
extern "C" void kernel_launch(void* const* d_in, const int* in_sizes, int n_in,
                              void* d_out, int out_size, void* d_ws, size_t ws_size,
                              hipStream_t stream) {
  (void)in_sizes; (void)n_in; (void)out_size;

  const float* x   = (const float*)d_in[0];
  const float* Wb  = (const float*)d_in[1];
  const float* bb  = (const float*)d_in[2];
  const float* A_s = (const float*)d_in[3];
  const float* B_s = (const float*)d_in[4];
  const float* A_r = (const float*)d_in[5];
  const float* B_r = (const float*)d_in[6];
  const float* Wr  = (const float*)d_in[7];
  const float* br  = (const float*)d_in[8];
  float* out = (float*)d_out;
  char* ws = (char*)d_ws;

  // ---- fast-path workspace layout (~162.5 MB) ----
  size_t off = 0;
  float* w_per_e = (float*)(ws + off);            off += (size_t)BS * 8 * 4;        // 512 KB
  float* hcat    = (float*)(ws + off);            off += (size_t)BS * NCAT * 4;     // 8 MB
  unsigned short* Acatt_hi = (unsigned short*)(ws + off); off += (size_t)NCAT * DD * 2;
  unsigned short* Acatt_lo = (unsigned short*)(ws + off); off += (size_t)NCAT * DD * 2;
  unsigned short* Bct_hi   = (unsigned short*)(ws + off); off += (size_t)DD * NCAT * 2;
  unsigned short* Bct_lo   = (unsigned short*)(ws + off); off += (size_t)DD * NCAT * 2;
  unsigned short* Wt_hi    = (unsigned short*)(ws + off); off += (size_t)DD * DD * 2;
  unsigned short* Wt_lo    = (unsigned short*)(ws + off); off += (size_t)DD * DD * 2;
  unsigned short* x_hi     = (unsigned short*)(ws + off); off += (size_t)BS * DD * 2;
  unsigned short* x_lo     = (unsigned short*)(ws + off); off += (size_t)BS * DD * 2;
  unsigned short* hcat_hi  = (unsigned short*)(ws + off); off += (size_t)BS * NCAT * 2;
  unsigned short* hcat_lo  = (unsigned short*)(ws + off); off += (size_t)BS * NCAT * 2;
  const size_t NEED_FAST = off;

  router_kernel<<<BS / 8, 256, 0, stream>>>(x, Wr, br, w_per_e);

  if (ws_size >= NEED_FAST) {
    // ---------------- fast path: pre-split bf16 + async-LDS GEMMs ----------
    split_f32_bf16<<<((size_t)BS * DD) / 256, 256, 0, stream>>>(x, x_hi, x_lo);
    transpose_split_2048<<<dim3(DD / 64, DD / 64), 256, 0, stream>>>(Wb, Wt_hi, Wt_lo);
    build_acatt_split<<<(NCAT * DD) / 256, 256, 0, stream>>>(A_s, A_r, Acatt_hi, Acatt_lo);
    build_bct_split<<<(DD * NCAT) / 256, 256, 0, stream>>>(B_s, B_r, Bct_hi, Bct_lo);

    // hcat = x . Acat (M=16384, N=128, K=2048)
    gemm_bf16_direct<<<dim3(BS / BM, NCAT / BN), 256, 0, stream>>>(
        x_hi, x_lo, Acatt_hi, Acatt_lo, DD,
        x_hi, x_lo, Acatt_hi, Acatt_lo, 0,
        nullptr, hcat, NCAT);

    fold_gates<<<(BS * NCAT) / 256, 256, 0, stream>>>(hcat, w_per_e);
    split_f32_bf16<<<((size_t)BS * NCAT) / 256, 256, 0, stream>>>(hcat, hcat_hi, hcat_lo);

    // out = x . W_base + hcat . Bcat + b_base   (fused K = 2048 + 128)
    gemm_bf16_direct<<<dim3(BS / BM, DD / BN), 256, 0, stream>>>(
        x_hi, x_lo, Wt_hi, Wt_lo, DD,
        hcat_hi, hcat_lo, Bct_hi, Bct_lo, NCAT,
        bb, out, DD);
  } else {
    // ---------------- fallback: split at staging, ~10.5 MB scratch ---------
    float* Acat = (float*)(ws + (size_t)BS * 8 * 4);
    float* Bcat = Acat + (size_t)DD * NCAT;
    float* hc   = Bcat + (size_t)NCAT * DD;

    build_acat<<<(DD * NCAT) / 256, 256, 0, stream>>>(A_s, A_r, Acat);
    build_bcat<<<(NCAT * DD) / 256, 256, 0, stream>>>(B_s, B_r, Bcat);

    gemm_bf16x3<<<dim3(BS / BM, NCAT / BN), 256, 0, stream>>>(
        x, DD, Acat, NCAT, DD,
        nullptr, 0, nullptr, 0, 0,
        nullptr, hc, NCAT);

    fold_gates<<<(BS * NCAT) / 256, 256, 0, stream>>>(hc, w_per_e);

    gemm_bf16x3<<<dim3(BS / BM, DD / BN), 256, 0, stream>>>(
        x, DD, Wb, DD, DD,
        hc, NCAT, Bcat, DD, NCAT,
        bb, out, DD);
  }
}
